// TorchLMHeadGRPO_13554916786396
// MI455X (gfx1250) — matrix operand order repaired and checked by reading, not compiled, not run on hardware
//
#include <hip/hip_runtime.h>

typedef __attribute__((ext_vector_type(16))) __bf16 v16bf;
typedef __attribute__((ext_vector_type(8)))  __bf16 v8bf;
typedef __attribute__((ext_vector_type(8)))  float  v8f;
typedef __attribute__((ext_vector_type(4)))  int    v4i;

#define B_  4
#define T_  1024
#define BT_ 4096      // B*T tokens
#define H_  2048
#define V_  32000
#define BM  256       // token tile
#define BN  128       // vocab tile
#define BK  32        // k chunk (one wmma depth)
#define BKP 40        // padded LDS row (80B stride: conflict-free, 16B aligned)
#define NKT (H_ / BK) // 64
#define NTHR 512      // 16 waves
#define BETA_     0.1f
#define EPS_LOW_  0.2f
#define EPS_HIGH_ 0.2f
#define INV_TEMP_ 1.0f

#define AS1 __attribute__((address_space(1)))
#define AS3 __attribute__((address_space(3)))

#if defined(__HIP_DEVICE_COMPILE__) && \
    __has_builtin(__builtin_amdgcn_global_load_async_to_lds_b128)
#define ASYNC_COPY 1
#else
#define ASYNC_COPY 0
#endif

// pack two f32 -> two bf16 (round-half-up) in ONE v_perm_b32 + two adds
__device__ __forceinline__ unsigned int pk2(float a, float b) {
    unsigned int ua = __float_as_uint(a) + 0x8000u;
    unsigned int ub = __float_as_uint(b) + 0x8000u;
    // result bytes: {a.b2, a.b3, b.b2, b.b3} -> lo16 = bf16(a), hi16 = bf16(b)
    return __builtin_amdgcn_perm(ub, ua, 0x07060302u);
}

__device__ __forceinline__ v16bf frag16(v8bf lo, v8bf hi) {
    return __builtin_shufflevector(lo, hi,
        0,1,2,3,4,5,6,7,8,9,10,11,12,13,14,15);
}

#if ASYNC_COPY
__device__ __forceinline__ void async_cp16(const void* g, void* l) {
    __builtin_amdgcn_global_load_async_to_lds_b128(
        (AS1 v4i*)g, (AS3 v4i*)l, 0, 0);
}
__device__ __forceinline__ void wait_async0() {
#if __has_builtin(__builtin_amdgcn_s_wait_asynccnt)
    __builtin_amdgcn_s_wait_asynccnt(0);
#else
    asm volatile("s_wait_asynccnt 0" ::: "memory");
#endif
}
#endif

// fp32 -> bf16 bulk pre-conversion (X / ref_input, done once)
__global__ __launch_bounds__(256)
void cvt_f32_bf16(const float4* __restrict__ src, uint2* __restrict__ dst, int n4) {
    int i = blockIdx.x * 256 + threadIdx.x;
    if (i < n4) {
        float4 f = src[i];
        dst[i] = make_uint2(pk2(f.x, f.y), pk2(f.z, f.w));
    }
}

__global__ void zero_ws(float* __restrict__ p, int n) {
    int i = blockIdx.x * 256 + threadIdx.x;
    if (i < n) p[i] = 0.0f;
}

// One (Xbf16, Wfp32) pair: sumexp[token] += sum_n exp(logit/T) over BN columns,
// sel[token] = logit[token, id]/T when id lands in this block's column range.
__global__ __launch_bounds__(NTHR)
void grpo_lse_gemm(const unsigned short* __restrict__ Xb, // bf16 [BT_, H_]
                   const float* __restrict__ W,           // fp32 [V_,  H_]
                   const int*   __restrict__ ids,         // [BT_]
                   float* __restrict__ sumexp,            // [BT_] pre-zeroed
                   float* __restrict__ sel)               // [BT_]
{
    __shared__ unsigned short lA[2][BM][BKP];  // 40KB (double-buffered)
    __shared__ unsigned short lB[2][BN][BKP];  // 20KB
    __shared__ float rowAcc[BM];
    __shared__ int   rowId[BM];

    const int tid  = threadIdx.x;
    const int lane = tid & 31;
    const int wave = tid >> 5;       // 0..15
    const int wr   = wave & 3;       // row group: rows wr*64 (4 tiles of 16)
    const int wcg  = wave >> 2;      // col group: cols wcg*32 (2 tiles of 16)
    const int half = lane >> 4;      // 0 | 1
    const int l16  = lane & 15;

    const int rowBlk = blockIdx.x * BM;  // token base (fast dim -> weight reuse)
    const int colBlk = blockIdx.y * BN;  // vocab base
    const int aRow   = wr * 64;
    const int cCol   = wcg * 32;

    if (tid < BM) { rowAcc[tid] = 0.0f; rowId[tid] = ids[rowBlk + tid]; }

    // A copy: 256x32 bf16 = 1024 x 16B chunks, 2 per thread
    const int ca0 = tid, ca1 = tid + NTHR;
    const int ar0 = ca0 >> 2, ao0 = (ca0 & 3) * 8;
    const int ar1 = ca1 >> 2, ao1 = (ca1 & 3) * 8;
    // B convert: 128x32 fp32 = 1024 float4, 2 per thread
    const int fb0 = tid, fb1 = tid + NTHR;
    const int br0 = fb0 >> 3, bo0 = (fb0 & 7) * 4;
    const int br1 = fb1 >> 3, bo1 = (fb1 & 7) * 4;

    v8f acc00 = {}, acc01 = {}, acc10 = {}, acc11 = {};
    v8f acc20 = {}, acc21 = {}, acc30 = {}, acc31 = {};

    // ---- prologue: fill buffer 0 with tile k=0 ----
    {
#if ASYNC_COPY
        async_cp16(&Xb[(size_t)(rowBlk + ar0) * H_ + ao0], &lA[0][ar0][ao0]);
        async_cp16(&Xb[(size_t)(rowBlk + ar1) * H_ + ao1], &lA[0][ar1][ao1]);
#else
        uint4 a0 = *(const uint4*)&Xb[(size_t)(rowBlk + ar0) * H_ + ao0];
        uint4 a1 = *(const uint4*)&Xb[(size_t)(rowBlk + ar1) * H_ + ao1];
        *(uint4*)&lA[0][ar0][ao0] = a0;
        *(uint4*)&lA[0][ar1][ao1] = a1;
#endif
        float4 b0 = *(const float4*)&W[(size_t)(colBlk + br0) * H_ + bo0];
        float4 b1 = *(const float4*)&W[(size_t)(colBlk + br1) * H_ + bo1];
        *(uint2*)&lB[0][br0][bo0] = make_uint2(pk2(b0.x, b0.y), pk2(b0.z, b0.w));
        *(uint2*)&lB[0][br1][bo1] = make_uint2(pk2(b1.x, b1.y), pk2(b1.z, b1.w));
#if ASYNC_COPY
        wait_async0();
#endif
        __syncthreads();
    }

    for (int kt = 0; kt < NKT; ++kt) {
        const int cur = kt & 1;
        const int nxt = cur ^ 1;
        const int k1  = (kt + 1) * BK;
        const bool havenext = (kt + 1) < NKT;  // uniform

        // ---- kick off next tile's A copy / B load (overlaps the WMMAs) ----
#if ASYNC_COPY
        if (havenext) {
            async_cp16(&Xb[(size_t)(rowBlk + ar0) * H_ + k1 + ao0], &lA[nxt][ar0][ao0]);
            async_cp16(&Xb[(size_t)(rowBlk + ar1) * H_ + k1 + ao1], &lA[nxt][ar1][ao1]);
        }
#else
        uint4 a0n = {}, a1n = {};
        if (havenext) {
            a0n = *(const uint4*)&Xb[(size_t)(rowBlk + ar0) * H_ + k1 + ao0];
            a1n = *(const uint4*)&Xb[(size_t)(rowBlk + ar1) * H_ + k1 + ao1];
        }
#endif
        float4 b0n = {}, b1n = {};
        if (havenext) {
            b0n = *(const float4*)&W[(size_t)(colBlk + br0) * H_ + k1 + bo0];
            b1n = *(const float4*)&W[(size_t)(colBlk + br1) * H_ + k1 + bo1];
            __builtin_prefetch(&W[(size_t)(colBlk + br0) * H_ + k1 + BK + bo0], 0, 1);
        }

        // ---- compute 8 WMMAs from buffer `cur` ----
        // B fragments (col n = l16): half 0 -> K0-7,16-23, half 1 -> K8-15,24-31
        const v8bf* bp0 = (const v8bf*)&lB[cur][cCol + l16][0];
        v16bf bf0 = frag16(bp0[half], bp0[2 + half]);
        const v8bf* bp1 = (const v8bf*)&lB[cur][cCol + 16 + l16][0];
        v16bf bf1 = frag16(bp1[half], bp1[2 + half]);

#define STEP(RT, A0, A1)                                                       \
        {                                                                      \
            const v8bf* ap = (const v8bf*)&lA[cur][aRow + (RT) * 16 + l16][0]; \
            v16bf af = frag16(ap[half], ap[2 + half]);                         \
            A0 = __builtin_amdgcn_wmma_f32_16x16x32_bf16(                      \
                false, af, false, bf0, (short)0, A0, false, false);            \
            A1 = __builtin_amdgcn_wmma_f32_16x16x32_bf16(                      \
                false, af, false, bf1, (short)0, A1, false, false);            \
        }
        STEP(0, acc00, acc01)
        STEP(1, acc10, acc11)
        STEP(2, acc20, acc21)
        STEP(3, acc30, acc31)
#undef STEP

        // ---- store next tile into the other buffer ----
        if (havenext) {
#if !ASYNC_COPY
            *(uint4*)&lA[nxt][ar0][ao0] = a0n;
            *(uint4*)&lA[nxt][ar1][ao1] = a1n;
#endif
            *(uint2*)&lB[nxt][br0][bo0] = make_uint2(pk2(b0n.x, b0n.y), pk2(b0n.z, b0n.w));
            *(uint2*)&lB[nxt][br1][bo1] = make_uint2(pk2(b1n.x, b1n.y), pk2(b1n.z, b1n.w));
        }
#if ASYNC_COPY
        wait_async0();
#endif
        __syncthreads();
    }

    // Epilogue: acc[j] = row (wr*64 + rt*16 + j + 8*half), col (cCol + ct*16 + l16)
#define EPI(ACC, RT, CT)                                                       \
    _Pragma("unroll")                                                          \
    for (int j = 0; j < 8; ++j) {                                              \
        const int rloc = aRow + (RT) * 16 + j + half * 8;                      \
        const float z = ACC[j] * INV_TEMP_;                                    \
        if (rowId[rloc] == colBlk + cCol + (CT) * 16 + l16)                    \
            sel[rowBlk + rloc] = z; /* unique writer grid-wide */              \
        float e = __expf(z);                                                   \
        e += __shfl_xor(e, 8, 16);                                             \
        e += __shfl_xor(e, 4, 16);                                             \
        e += __shfl_xor(e, 2, 16);                                             \
        e += __shfl_xor(e, 1, 16);                                             \
        if (l16 == 0) atomicAdd(&rowAcc[rloc], e);                             \
    }
    EPI(acc00, 0, 0) EPI(acc01, 0, 1)
    EPI(acc10, 1, 0) EPI(acc11, 1, 1)
    EPI(acc20, 2, 0) EPI(acc21, 2, 1)
    EPI(acc30, 3, 0) EPI(acc31, 3, 1)
#undef EPI
    __syncthreads();
    if (tid < BM) atomicAdd(&sumexp[rowBlk + tid], rowAcc[tid]);
}

__global__ __launch_bounds__(1024)
void grpo_reduce(const float* __restrict__ se1, const float* __restrict__ sel1,
                 const float* __restrict__ se2, const float* __restrict__ sel2,
                 const float* __restrict__ adv, const float* __restrict__ mask,
                 float* __restrict__ out)
{
    __shared__ float accS[4];  // loss, kl, clip, mask_sum
    if (threadIdx.x < 4) accS[threadIdx.x] = 0.0f;
    __syncthreads();

    float loss = 0.f, klm = 0.f, clipm = 0.f, ms = 0.f;
    for (int t = threadIdx.x; t < BT_; t += 1024) {
        const float lp  = sel1[t] - __logf(se1[t]);
        const float rlp = sel2[t] - __logf(se2[t]);
        const float m   = mask[t];
        const float a   = adv[t / T_];
        const float coef1 = __expf(lp - lp);   // stop_grad(lp)==lp -> 1.0
        const float coef2 = fminf(fmaxf(coef1, 1.0f - EPS_LOW_), 1.0f + EPS_HIGH_);
        float ptl = -fminf(coef1 * a, coef2 * a);
        const float d  = rlp - lp;
        const float kl = __expf(d) - d - 1.0f;
        ptl += BETA_ * kl;
        loss  += ptl * m;
        klm   += kl * m;
        const bool clip = ((coef1 < 1.0f - EPS_LOW_) && (a < 0.f)) ||
                          ((coef1 > 1.0f + EPS_HIGH_) && (a > 0.f));
        clipm += (clip ? 1.0f : 0.0f) * m;
        ms    += m;
    }
    #pragma unroll
    for (int off = 16; off >= 1; off >>= 1) {
        loss  += __shfl_xor(loss,  off, 32);
        klm   += __shfl_xor(klm,   off, 32);
        clipm += __shfl_xor(clipm, off, 32);
        ms    += __shfl_xor(ms,    off, 32);
    }
    if ((threadIdx.x & 31) == 0) {
        atomicAdd(&accS[0], loss);
        atomicAdd(&accS[1], klm);
        atomicAdd(&accS[2], clipm);
        atomicAdd(&accS[3], ms);
    }
    __syncthreads();
    if (threadIdx.x == 0) {
        const float msum = fmaxf(accS[3], 1.0f);
        out[0] = accS[0] / msum;
        out[1] = accS[1] / msum;
        out[2] = accS[2] / msum;
    }
}

extern "C" void kernel_launch(void* const* d_in, const int* in_sizes, int n_in,
                              void* d_out, int out_size, void* d_ws, size_t ws_size,
                              hipStream_t stream) {
    const float* x      = (const float*)d_in[0];
    const float* ref_x  = (const float*)d_in[1];
    const float* w      = (const float*)d_in[2];
    const float* ref_w  = (const float*)d_in[3];
    const float* adv    = (const float*)d_in[4];
    const float* mask   = (const float*)d_in[5];
    const int*   ids    = (const int*)d_in[6];

    float* ws   = (float*)d_ws;
    float* se1  = ws;            // [BT_]
    float* se2  = ws + BT_;      // [BT_]
    float* sel1 = ws + 2 * BT_;  // [BT_]
    float* sel2 = ws + 3 * BT_;  // [BT_]
    unsigned short* xb1 = (unsigned short*)(ws + 4 * BT_);  // bf16 X    [BT_*H_]
    unsigned short* xb2 = xb1 + (size_t)BT_ * H_;           // bf16 refX [BT_*H_]

    zero_ws<<<(2 * BT_ + 255) / 256, 256, 0, stream>>>(ws, 2 * BT_);

    const int n4 = BT_ * H_ / 4;
    cvt_f32_bf16<<<(n4 + 255) / 256, 256, 0, stream>>>(
        (const float4*)x,     (uint2*)xb1, n4);
    cvt_f32_bf16<<<(n4 + 255) / 256, 256, 0, stream>>>(
        (const float4*)ref_x, (uint2*)xb2, n4);

    dim3 grid(BT_ / BM, V_ / BN);  // x = token tiles (fast) -> weights stream once
    grpo_lse_gemm<<<grid, NTHR, 0, stream>>>(xb1, w,     ids, se1, sel1);
    grpo_lse_gemm<<<grid, NTHR, 0, stream>>>(xb2, ref_w, ids, se2, sel2);

    grpo_reduce<<<1, 1024, 0, stream>>>(se1, sel1, se2, sel2, adv, mask,
                                        (float*)d_out);
}
